// Model_3839700763122
// MI455X (gfx1250) — matrix operand order repaired and checked by reading, not compiled, not run on hardware
//
#include <hip/hip_runtime.h>
#include <hip/hip_bf16.h>
#include <math.h>

// ---------------------------------------------------------------------------
// Autoformer forward on MI455X (gfx1250, wave32, WMMA).
// Big matmuls: bf16 WMMA GEMM (v_wmma_f32_16x16x32_bf16), fp32 accumulate,
// float4 global staging software-pipelined against the WMMA inner loop,
// fp32->bf16x2 packing done with a single v_cvt_pk_bf16_f32 per dword.
// FFT autocorrelation reduced to batched Q@K^T (WMMA) + coalesced circulant
// diagonal sums, since the reference only consumes corr.mean(axis=(H,E)).
// ---------------------------------------------------------------------------

#define NB     16
#define LSEQ   512
#define DMODEL 512
#define CIN    7
#define MARKD  4
#define DFF    2048
#define TOPK   18              // int(3*ln(512)) = 18
#define MROWS  (NB * LSEQ)     // 8192
#define CORR_CH 4              // row chunks for diag-sum partials

typedef __attribute__((ext_vector_type(16))) __bf16 v16bf;
typedef __attribute__((ext_vector_type(8)))  float  v8f;

static inline int ceildiv_i(long long a, int b) { return (int)((a + b - 1) / b); }

// Pack two f32 into one dword of bf16x2 with the native packed convert.
// Non-volatile: pure op, lets the scheduler pick registers and clause the
// subsequent ds_store_b32s instead of serializing on one destination VGPR.
static __device__ __forceinline__ unsigned pack_bf16(float a, float b)
{
    unsigned r;
    asm("v_cvt_pk_bf16_f32 %0, %1, %2" : "=v"(r) : "v"(a), "v"(b));
    return r;
}

// ---------------------------------------------------------------------------
// WMMA GEMM: C[M,N] = epi(A[M,K] @ W[N,K]^T + bias + R), fp32 in/out, bf16 MACs.
// Block tile 128x64, BK=32, 8 waves; each wave owns a 32x32 tile (2x2 WMMA).
// Staging: float4 global loads (b128 clauses), next tile kept in registers
// while current tile is consumed from LDS (software pipeline).
// Batched via blockIdx.z with strides sA/sW/sC (0 => shared operand).
// All M,N,K used here are multiples of 128/64/32 -> no bounds checks.
// ---------------------------------------------------------------------------
__global__ __launch_bounds__(256) void k_gemm_bf16(
    const float* __restrict__ A, const float* __restrict__ Wt,
    const float* __restrict__ bias, const float* __restrict__ R,
    float* __restrict__ C, int M, int N, int K,
    long long sA, long long sW, long long sC, int gelu)
{
    A  += (long long)blockIdx.z * sA;
    Wt += (long long)blockIdx.z * sW;
    C  += (long long)blockIdx.z * sC;
    if (R) R += (long long)blockIdx.z * sC;

    __shared__ __bf16 As[128][34];   // +2 halves pad; rows stay 4B aligned
    __shared__ __bf16 Bs[64][34];

    const int tid  = threadIdx.x;
    const int lane = tid & 31;
    const int wave = tid >> 5;
    const int wm   = wave >> 1;            // 0..3 (rows)
    const int wn   = wave & 1;             // 0..1 (cols)
    const int row0 = blockIdx.y * 128;
    const int col0 = blockIdx.x * 64;

    // float4 staging assignments: 8 float4 per 32-float row
    int arw[4], acl[4], brw[2], bcl[2];
#pragma unroll
    for (int i = 0; i < 4; ++i) { int idx = tid + i * 256; arw[i] = idx >> 3; acl[i] = (idx & 7) * 4; }
#pragma unroll
    for (int i = 0; i < 2; ++i) { int idx = tid + i * 256; brw[i] = idx >> 3; bcl[i] = (idx & 7) * 4; }

    const float* aPtr[4]; const float* bPtr[2];
#pragma unroll
    for (int i = 0; i < 4; ++i) aPtr[i] = A  + (long long)(row0 + arw[i]) * K + acl[i];
#pragma unroll
    for (int i = 0; i < 2; ++i) bPtr[i] = Wt + (long long)(col0 + brw[i]) * K + bcl[i];

    v8f acc[2][2] = {};

    float4 areg[4], breg[2];
#pragma unroll
    for (int i = 0; i < 4; ++i) areg[i] = *(const float4*)(aPtr[i]);
#pragma unroll
    for (int i = 0; i < 2; ++i) breg[i] = *(const float4*)(bPtr[i]);

    for (int k0 = 0; k0 < K; k0 += 32) {
        // stage held registers -> LDS as packed bf16 (dword stores)
#pragma unroll
        for (int i = 0; i < 4; ++i) {
            unsigned* d = (unsigned*)&As[arw[i]][acl[i]];
            d[0] = pack_bf16(areg[i].x, areg[i].y);
            d[1] = pack_bf16(areg[i].z, areg[i].w);
        }
#pragma unroll
        for (int i = 0; i < 2; ++i) {
            unsigned* d = (unsigned*)&Bs[brw[i]][bcl[i]];
            d[0] = pack_bf16(breg[i].x, breg[i].y);
            d[1] = pack_bf16(breg[i].z, breg[i].w);
        }
        __syncthreads();

        // issue next K-tile global loads early; latency hides behind WMMAs
        const bool more = (k0 + 32) < K;
        if (more) {
#pragma unroll
            for (int i = 0; i < 4; ++i) areg[i] = *(const float4*)(aPtr[i] + k0 + 32);
#pragma unroll
            for (int i = 0; i < 2; ++i) breg[i] = *(const float4*)(bPtr[i] + k0 + 32);
            if (k0 + 64 < K) {   // pull tile-after-next toward L2 (global_prefetch_b8)
                __builtin_prefetch(aPtr[0] + k0 + 64, 0, 1);
                __builtin_prefetch(bPtr[0] + k0 + 64, 0, 1);
            }
        }

        const int lm   = lane & 15;
        const int hi   = lane >> 4;
        const int kb   = hi << 3;   // A frag: ISA 16-bit A 16x32 layout
        const int koff = hi << 4;   // unused: B frag uses same k order as A (kb)

        v16bf af[2], bfr[2];
#pragma unroll
        for (int s = 0; s < 2; ++s) {
            const __bf16* ap = &As[wm * 32 + s * 16 + lm][0];
            const __bf16* bp = &Bs[wn * 32 + s * 16 + lm][0];
#pragma unroll
            for (int e = 0; e < 8; ++e) {
                af[s][e]     = ap[kb + e];
                af[s][e + 8] = ap[16 + kb + e];
            }
#pragma unroll
            for (int e = 0; e < 16; ++e)
                bfr[s][e] = bp[((e & 8) << 1) + kb + (e & 7)];
        }
#pragma unroll
        for (int sm = 0; sm < 2; ++sm)
#pragma unroll
            for (int sn = 0; sn < 2; ++sn)
                acc[sm][sn] = __builtin_amdgcn_wmma_f32_16x16x32_bf16(
                    false, af[sm], false, bfr[sn], (short)0, acc[sm][sn],
                    false, false);
        __syncthreads();
    }

    // epilogue: C/D layout = lane(n = lane&15), VGPR r -> m = r + 8*(lane>=16)
    const int lm = lane & 15;
    const int hi = lane >> 4;
#pragma unroll
    for (int sm = 0; sm < 2; ++sm) {
        int rowb = row0 + wm * 32 + sm * 16 + hi * 8;
#pragma unroll
        for (int sn = 0; sn < 2; ++sn) {
            int col = col0 + wn * 32 + sn * 16 + lm;
            float bv = bias ? bias[col] : 0.0f;
#pragma unroll
            for (int r = 0; r < 8; ++r) {
                long long o = (long long)(rowb + r) * N + col;
                float v = acc[sm][sn][r] + bv;
                if (gelu) v = 0.5f * v * (1.0f + erff(v * 0.70710678118654752f));
                if (R) v += R[o];
                C[o] = v;
            }
        }
    }
}

// --------------------------- elementwise kernels ---------------------------

// series_decomp, generic channel count (used for CIN=7 inputs).
__global__ void k_decomp(const float* __restrict__ x, float* __restrict__ seas,
                         float* __restrict__ trend, int accum, int Cc, long long n)
{
    long long i = (long long)blockIdx.x * blockDim.x + threadIdx.x;
    if (i >= n) return;
    int c = (int)(i % Cc);
    int l = (int)((i / Cc) % LSEQ);
    long long b = i / ((long long)Cc * LSEQ);
    const float* xb = x + b * (long long)LSEQ * Cc + c;
    float s = 0.f;
#pragma unroll
    for (int j = -12; j <= 12; ++j) {
        int t = l + j;
        t = t < 0 ? 0 : (t > LSEQ - 1 ? LSEQ - 1 : t);
        s += xb[(long long)t * Cc];
    }
    float mm = s * (1.0f / 25.0f);
    if (seas)  seas[i] = x[i] - mm;
    if (trend) { if (accum) trend[i] += mm; else trend[i] = mm; }
}

// series_decomp specialized for D=512 activations, float4 wide.
__global__ void k_decomp512(const float4* __restrict__ x, float4* __restrict__ seas,
                            float4* __restrict__ trend, int accum)
{
    long long i = (long long)blockIdx.x * blockDim.x + threadIdx.x;
    if (i >= (long long)MROWS * 128) return;
    int c4 = (int)(i & 127);
    int l  = (int)((i >> 7) & (LSEQ - 1));
    long long b = i >> 16;
    const float4* xb = x + b * (long long)LSEQ * 128 + c4;
    float sx = 0.f, sy = 0.f, sz = 0.f, sw = 0.f;
#pragma unroll
    for (int j = -12; j <= 12; ++j) {
        int t = l + j;
        t = t < 0 ? 0 : (t > LSEQ - 1 ? LSEQ - 1 : t);
        float4 v = xb[(long long)t * 128];
        sx += v.x; sy += v.y; sz += v.z; sw += v.w;
    }
    const float r = 1.0f / 25.0f;
    float4 mm = make_float4(sx * r, sy * r, sz * r, sw * r);
    if (seas) {
        float4 xi = x[i];
        seas[i] = make_float4(xi.x - mm.x, xi.y - mm.y, xi.z - mm.z, xi.w - mm.w);
    }
    if (trend) {
        if (accum) {
            float4 t0 = trend[i];
            trend[i] = make_float4(t0.x + mm.x, t0.y + mm.y, t0.z + mm.z, t0.w + mm.w);
        } else {
            trend[i] = mm;
        }
    }
}

__global__ void k_mean_enc(const float* __restrict__ x, float* __restrict__ mean)
{
    int i = blockIdx.x * blockDim.x + threadIdx.x;
    if (i >= NB * CIN) return;
    int c = i % CIN, b = i / CIN;
    const float* xb = x + (long long)b * LSEQ * CIN + c;
    float s = 0.f;
    for (int l = 0; l < LSEQ; ++l) s += xb[l * CIN];
    mean[i] = s * (1.0f / LSEQ);
}

__global__ void k_seeds(const float* __restrict__ seas_raw, const float* __restrict__ trend_raw,
                        const float* __restrict__ mean, float* __restrict__ seas_init,
                        float* __restrict__ trend_tot)
{
    int i = blockIdx.x * blockDim.x + threadIdx.x;
    if (i >= NB * LSEQ * CIN) return;
    int c = i % CIN;
    int l = (i / CIN) % LSEQ;
    int b = i / (CIN * LSEQ);
    if (l < 256) {
        int src = (b * LSEQ + 256 + l) * CIN + c;
        seas_init[i] = seas_raw[src];
        trend_tot[i] = trend_raw[src];
    } else {
        seas_init[i] = 0.f;
        trend_tot[i] = mean[b * CIN + c];
    }
}

// Embedding: circular conv3 over channels (7->512) + mark linear (4->512).
__global__ void k_embed(const float* __restrict__ x, const float* __restrict__ mark,
                        const float* __restrict__ Wv3, const float* __restrict__ Wm,
                        float* __restrict__ out)
{
    long long i = (long long)blockIdx.x * blockDim.x + threadIdx.x;
    if (i >= (long long)MROWS * DMODEL) return;
    int d = (int)(i & (DMODEL - 1));
    int l = (int)((i >> 9) & (LSEQ - 1));
    int b = (int)(i >> 18);
    float acc = 0.f;
#pragma unroll
    for (int k = 0; k < 3; ++k) {
        int ls = (l + k - 1 + LSEQ) & (LSEQ - 1);   // circular pad
        const float* xr = x + ((long long)b * LSEQ + ls) * CIN;
        const float* wr = Wv3 + (long long)d * CIN * 3 + k;
#pragma unroll
        for (int c = 0; c < CIN; ++c) acc += xr[c] * wr[c * 3];
    }
    const float* mr = mark + ((long long)b * LSEQ + l) * MARKD;
    const float* wm = Wm + (long long)d * MARKD;
#pragma unroll
    for (int m = 0; m < MARKD; ++m) acc += mr[m] * wm[m];
    out[i] = acc;
}

// Circulant diagonal partial sums, coalesced:
// corr[b,tau] = sum_i S[b, i, (i - tau) & 511]  (row reads are contiguous)
__global__ __launch_bounds__(512) void k_corr_part(const float* __restrict__ S,
                                                   float* __restrict__ part)
{
    int b = blockIdx.x, ch = blockIdx.y;
    int tau = threadIdx.x;
    const float* Sb = S + (long long)b * LSEQ * LSEQ;
    float s = 0.f;
    const int i0 = ch * (LSEQ / CORR_CH);
    for (int i = i0; i < i0 + LSEQ / CORR_CH; ++i)
        s += Sb[(long long)i * LSEQ + ((i - tau) & (LSEQ - 1))];
    part[((long long)b * CORR_CH + ch) * LSEQ + tau] = s;
}

__global__ void k_corr_reduce(const float* __restrict__ part, float* __restrict__ corr)
{
    int i = blockIdx.x * blockDim.x + threadIdx.x;
    if (i >= NB * LSEQ) return;
    int b = i >> 9, tau = i & (LSEQ - 1);
    float s = 0.f;
#pragma unroll
    for (int ch = 0; ch < CORR_CH; ++ch)
        s += part[((long long)b * CORR_CH + ch) * LSEQ + tau];
    corr[i] = s * (1.0f / DMODEL);
}

// top-18 + softmax, one wave32 per batch: 16 values/lane, wave argmax rounds.
__global__ void k_topk_softmax(const float* __restrict__ corr, float* __restrict__ w,
                               int* __restrict__ dly)
{
    int b = blockIdx.x;
    int lane = threadIdx.x;   // 32
    const float* c = corr + b * LSEQ;
    float v[16];
#pragma unroll
    for (int j = 0; j < 16; ++j) v[j] = c[lane * 16 + j];

    __shared__ float svals[TOPK];
    __shared__ int   sidx[TOPK];
    for (int k = 0; k < TOPK; ++k) {
        float best = -3.4e38f; int bj = 0;
#pragma unroll
        for (int j = 0; j < 16; ++j)
            if (v[j] > best) { best = v[j]; bj = j; }
        int bidx = lane * 16 + bj;
#pragma unroll
        for (int off = 16; off > 0; off >>= 1) {
            float ov = __shfl_xor(best, off, 32);
            int   oi = __shfl_xor(bidx, off, 32);
            if (ov > best || (ov == best && oi < bidx)) { best = ov; bidx = oi; }
        }
        if (lane == (bidx >> 4)) v[bidx & 15] = -3.4e38f;   // knock out winner
        if (lane == 0) { svals[k] = best; sidx[k] = bidx; }
    }
    __syncthreads();
    if (lane == 0) {
        float m = svals[0];
        for (int k = 1; k < TOPK; ++k) m = fmaxf(m, svals[k]);
        float s = 0.f, e[TOPK];
        for (int k = 0; k < TOPK; ++k) { e[k] = expf(svals[k] - m); s += e[k]; }
        float inv = 1.0f / s;
        for (int k = 0; k < TOPK; ++k) {
            w[b * TOPK + k]   = e[k] * inv;
            dly[b * TOPK + k] = sidx[k];
        }
    }
}

// out[b,l,d] = sum_k w[b,k] * V[b, (l+delay[b,k])%L, d]   (float4 wide)
__global__ void k_agg(const float4* __restrict__ V, const float* __restrict__ w,
                      const int* __restrict__ dly, float4* __restrict__ out)
{
    long long i = (long long)blockIdx.x * blockDim.x + threadIdx.x;
    if (i >= (long long)MROWS * 128) return;
    int d4 = (int)(i & 127);
    int l  = (int)((i >> 7) & (LSEQ - 1));
    int b  = (int)(i >> 16);
    const float4* Vb = V + (long long)b * LSEQ * 128;
    float ax = 0.f, ay = 0.f, az = 0.f, aw = 0.f;
#pragma unroll
    for (int k = 0; k < TOPK; ++k) {
        int   dl = (l + dly[b * TOPK + k]) & (LSEQ - 1);
        float wk = w[b * TOPK + k];
        float4 t = Vb[(long long)dl * 128 + d4];
        ax += wk * t.x; ay += wk * t.y; az += wk * t.z; aw += wk * t.w;
    }
    out[i] = make_float4(ax, ay, az, aw);
}

// _my_layernorm stage 1: per-row (b,l) normalize over D, scale+shift.
__global__ __launch_bounds__(256) void k_ln1(const float* __restrict__ x,
                                             const float* __restrict__ g,
                                             const float* __restrict__ bb,
                                             float* __restrict__ xh)
{
    int row = blockIdx.x;
    int tid = threadIdx.x;
    const float* xr = x + (long long)row * DMODEL;
    float v0 = xr[tid], v1 = xr[tid + 256];
    __shared__ float s1[256], s2[256];
    s1[tid] = v0 + v1;
    s2[tid] = v0 * v0 + v1 * v1;
    __syncthreads();
    for (int off = 128; off > 0; off >>= 1) {
        if (tid < off) { s1[tid] += s1[tid + off]; s2[tid] += s2[tid + off]; }
        __syncthreads();
    }
    float mu  = s1[0] * (1.0f / DMODEL);
    float var = s2[0] * (1.0f / DMODEL) - mu * mu;
    float inv = rsqrtf(var + 1e-5f);
    xh[(long long)row * DMODEL + tid]       = (v0 - mu) * inv * g[tid] + bb[tid];
    xh[(long long)row * DMODEL + tid + 256] = (v1 - mu) * inv * g[tid + 256] + bb[tid + 256];
}

// stage 2a: column means over L per (b,d)
__global__ void k_colmean(const float* __restrict__ xh, float* __restrict__ cm)
{
    int i = blockIdx.x * blockDim.x + threadIdx.x;
    if (i >= NB * DMODEL) return;
    int d = i & (DMODEL - 1);
    int b = i >> 9;
    const float* p = xh + (long long)b * LSEQ * DMODEL + d;
    float s = 0.f;
    for (int l = 0; l < LSEQ; ++l) s += p[(long long)l * DMODEL];
    cm[i] = s * (1.0f / LSEQ);
}

// stage 2b: subtract column mean (float4 wide)
__global__ void k_subcol(const float4* __restrict__ xh, const float* __restrict__ cm,
                         float4* __restrict__ out)
{
    long long i = (long long)blockIdx.x * blockDim.x + threadIdx.x;
    if (i >= (long long)MROWS * 128) return;
    int d4 = (int)(i & 127);
    int b  = (int)(i >> 16);
    float4 v  = xh[i];
    float4 cv = *(const float4*)(cm + (long long)b * DMODEL + d4 * 4);
    out[i] = make_float4(v.x - cv.x, v.y - cv.y, v.z - cv.z, v.w - cv.w);
}

// trend_tot += circ_conv3(tacc, Wtp)   (D=512 -> C=7)
__global__ void k_trend_conv(const float* __restrict__ t, const float* __restrict__ Wtp,
                             float* __restrict__ trend_tot)
{
    int i = blockIdx.x * blockDim.x + threadIdx.x;
    if (i >= NB * LSEQ * CIN) return;
    int c = i % CIN;
    int l = (i / CIN) % LSEQ;
    int b = i / (CIN * LSEQ);
    const float* tb = t + (long long)b * LSEQ * DMODEL;
    float acc = 0.f;
    for (int k = 0; k < 3; ++k) {
        int ls = (l + k - 1 + LSEQ) & (LSEQ - 1);
        const float* tr = tb + (long long)ls * DMODEL;
        const float* w  = Wtp + (long long)c * DMODEL * 3 + k;
        for (int d = 0; d < DMODEL; ++d) acc += tr[d] * w[d * 3];
    }
    trend_tot[i] += acc;
}

// final: out[b,p,c] = trend_tot[b,256+p,c] + proj(dec_ln[b,256+p,:])
__global__ void k_final(const float* __restrict__ decln, const float* __restrict__ Wp,
                        const float* __restrict__ bp, const float* __restrict__ trend_tot,
                        float* __restrict__ out)
{
    int i = blockIdx.x * blockDim.x + threadIdx.x;
    if (i >= NB * 256 * CIN) return;
    int c = i % CIN;
    int p = (i / CIN) % 256;
    int b = i / (CIN * 256);
    int l = 256 + p;
    const float* xr = decln + ((long long)b * LSEQ + l) * DMODEL;
    const float* wr = Wp + (long long)c * DMODEL;
    float acc = bp[c];
    for (int d = 0; d < DMODEL; ++d) acc += xr[d] * wr[d];
    out[i] = acc + trend_tot[(b * LSEQ + l) * CIN + c];
}

// ------------------------------- host side ---------------------------------

static void run_gemm(hipStream_t st, const float* A, const float* W, const float* bias,
                     const float* R, float* C, int M, int N, int K,
                     long long sA, long long sW, long long sC, int batches, int gelu)
{
    dim3 g(N / 64, M / 128, batches);
    k_gemm_bf16<<<g, dim3(256), 0, st>>>(A, W, bias, R, C, M, N, K, sA, sW, sC, gelu);
}

static void run_decomp512(hipStream_t st, const float* x, float* seas, float* trend, int accum)
{
    long long n = (long long)MROWS * 128;
    k_decomp512<<<ceildiv_i(n, 256), 256, 0, st>>>(
        (const float4*)x, (float4*)seas, (float4*)trend, accum);
}

struct AttnW { const float *Wq, *bq, *Wk, *bk, *Wv, *bv, *Wo, *bo; };

// out = q_in + proj(autocorr(q_in, kv_in));  trashes qb,kbuf,vbuf,Sb,aggout
static void run_attn(hipStream_t st, const float* q_in, const float* kv_in, const AttnW& w,
                     float* qb, float* kbuf, float* vbuf, float* Sb, float* part,
                     float* corr, float* wts, int* dly, float* aggout, float* out)
{
    run_gemm(st, q_in,  w.Wq, w.bq, nullptr, qb,   MROWS, DMODEL, DMODEL, 0, 0, 0, 1, 0);
    run_gemm(st, kv_in, w.Wk, w.bk, nullptr, kbuf, MROWS, DMODEL, DMODEL, 0, 0, 0, 1, 0);
    run_gemm(st, kv_in, w.Wv, w.bv, nullptr, vbuf, MROWS, DMODEL, DMODEL, 0, 0, 0, 1, 0);
    const long long sQ = (long long)LSEQ * DMODEL;   // per-batch Q/K stride
    const long long sS = (long long)LSEQ * LSEQ;     // per-batch score stride
    run_gemm(st, qb, kbuf, nullptr, nullptr, Sb, LSEQ, LSEQ, DMODEL, sQ, sQ, sS, NB, 0);
    k_corr_part<<<dim3(NB, CORR_CH), 512, 0, st>>>(Sb, part);
    k_corr_reduce<<<ceildiv_i(NB * LSEQ, 256), 256, 0, st>>>(part, corr);
    k_topk_softmax<<<NB, 32, 0, st>>>(corr, wts, dly);
    long long n4 = (long long)MROWS * 128;
    k_agg<<<ceildiv_i(n4, 256), 256, 0, st>>>((const float4*)vbuf, wts, dly, (float4*)aggout);
    run_gemm(st, aggout, w.Wo, w.bo, /*residual=*/q_in, out, MROWS, DMODEL, DMODEL, 0, 0, 0, 1, 0);
}

extern "C" void kernel_launch(void* const* d_in, const int* in_sizes, int n_in,
                              void* d_out, int out_size, void* d_ws, size_t ws_size,
                              hipStream_t stream)
{
    (void)in_sizes; (void)n_in; (void)out_size; (void)ws_size;

    // ---- inputs (jax sorted-dict flatten order) ----
    const float* x_enc      = (const float*)d_in[0];
    const float* x_mark_enc = (const float*)d_in[1];
    /* d_in[2] = x_dec: values never used by the reference */
    const float* x_mark_dec = (const float*)d_in[3];
    const float* dec_Wm  = (const float*)d_in[4];
    const float* dec_Wv3 = (const float*)d_in[5];
    const float* dWc1 = (const float*)d_in[6];
    const float* dWc2 = (const float*)d_in[7];
    const float* dWtp = (const float*)d_in[8];
    AttnW cross = { (const float*)d_in[11], (const float*)d_in[15],   // Wq, bq
                    (const float*)d_in[9],  (const float*)d_in[13],   // Wk, bk
                    (const float*)d_in[12], (const float*)d_in[16],   // Wv, bv
                    (const float*)d_in[10], (const float*)d_in[14] }; // Wo, bo
    AttnW dself = { (const float*)d_in[19], (const float*)d_in[23],
                    (const float*)d_in[17], (const float*)d_in[21],
                    (const float*)d_in[20], (const float*)d_in[24],
                    (const float*)d_in[18], (const float*)d_in[22] };
    const float* dn_b = (const float*)d_in[25];
    const float* dn_g = (const float*)d_in[26];
    const float* enc_Wm  = (const float*)d_in[27];
    const float* enc_Wv3 = (const float*)d_in[28];
    struct { const float *Wc1, *Wc2; AttnW a; } ep[2];
    for (int li = 0; li < 2; ++li) {
        int o = 29 + li * 10;
        ep[li].Wc1 = (const float*)d_in[o + 0];
        ep[li].Wc2 = (const float*)d_in[o + 1];
        ep[li].a   = { (const float*)d_in[o + 4], (const float*)d_in[o + 8],
                       (const float*)d_in[o + 2], (const float*)d_in[o + 6],
                       (const float*)d_in[o + 5], (const float*)d_in[o + 9],
                       (const float*)d_in[o + 3], (const float*)d_in[o + 7] };
    }
    const float* en_b = (const float*)d_in[49];
    const float* en_g = (const float*)d_in[50];
    const float* pW   = (const float*)d_in[51];
    const float* pb   = (const float*)d_in[52];

    // ---- workspace carve (fp32) ----
    const size_t MD = (size_t)MROWS * DMODEL;     // 4.19M floats
    float* p = (float*)d_ws;
    float* enc  = p; p += MD;
    float* dec  = p; p += MD;
    float* qb   = p; p += MD;
    float* kbuf = p; p += MD;
    float* vbuf = p; p += MD;
    float* t1   = p; p += MD;
    float* t2   = p; p += MD;
    float* tacc = p; p += MD;
    float* hbuf = p; p += (size_t)MROWS * DFF;    // FFN hidden; also aliased as S
    float* Sb   = hbuf;                           // scores never live with hbuf
    float* part = p; p += NB * CORR_CH * LSEQ;
    float* corr = p; p += NB * LSEQ;
    float* wts  = p; p += NB * TOPK;
    int*   dly  = (int*)p; p += NB * TOPK;
    float* menc = p; p += NB * CIN;
    float* seas_raw  = p; p += NB * LSEQ * CIN;
    float* trend_raw = p; p += NB * LSEQ * CIN;
    float* seas_init = p; p += NB * LSEQ * CIN;
    float* trend_tot = p; p += NB * LSEQ * CIN;
    float* cm = p; p += NB * DMODEL;

    const long long n7  = (long long)NB * LSEQ * CIN;
    const long long nMD = (long long)MROWS * DMODEL;

    // ---- prelude: decomp of x_enc, decoder seeds ----
    k_decomp<<<ceildiv_i(n7, 256), 256, 0, stream>>>(x_enc, seas_raw, trend_raw, 0, CIN, n7);
    k_mean_enc<<<1, 128, 0, stream>>>(x_enc, menc);
    k_seeds<<<ceildiv_i(n7, 256), 256, 0, stream>>>(seas_raw, trend_raw, menc, seas_init, trend_tot);

    // ---- encoder ----
    k_embed<<<ceildiv_i(nMD, 256), 256, 0, stream>>>(x_enc, x_mark_enc, enc_Wv3, enc_Wm, enc);
    for (int li = 0; li < 2; ++li) {
        run_attn(stream, enc, enc, ep[li].a, qb, kbuf, vbuf, Sb, part, corr, wts, dly, t1, t2);
        run_decomp512(stream, t2, t1, nullptr, 0);
        run_gemm(stream, t1, ep[li].Wc1, nullptr, nullptr, hbuf, MROWS, DFF, DMODEL, 0, 0, 0, 1, 1);
        run_gemm(stream, hbuf, ep[li].Wc2, nullptr, t1, t2, MROWS, DMODEL, DFF, 0, 0, 0, 1, 0);
        run_decomp512(stream, t2, enc, nullptr, 0);
    }
    // encoder _my_layernorm -> enc
    k_ln1<<<MROWS, 256, 0, stream>>>(enc, en_g, en_b, t2);
    k_colmean<<<ceildiv_i(NB * DMODEL, 256), 256, 0, stream>>>(t2, cm);
    k_subcol<<<ceildiv_i((long long)MROWS * 128, 256), 256, 0, stream>>>(
        (const float4*)t2, cm, (float4*)enc);

    // ---- decoder ----
    k_embed<<<ceildiv_i(nMD, 256), 256, 0, stream>>>(seas_init, x_mark_dec, dec_Wv3, dec_Wm, dec);

    run_attn(stream, dec, dec, dself, qb, kbuf, vbuf, Sb, part, corr, wts, dly, t1, t2);
    run_decomp512(stream, t2, t1, tacc, 0);                                   // x, t1
    run_attn(stream, t1, enc, cross, qb, kbuf, vbuf, Sb, part, corr, wts, dly, dec, t2);
    run_decomp512(stream, t2, t1, tacc, 1);                                   // x, t2
    run_gemm(stream, t1, dWc1, nullptr, nullptr, hbuf, MROWS, DFF, DMODEL, 0, 0, 0, 1, 1);
    run_gemm(stream, hbuf, dWc2, nullptr, t1, t2, MROWS, DMODEL, DFF, 0, 0, 0, 1, 0);
    run_decomp512(stream, t2, dec, tacc, 1);                                  // x, t3
    k_trend_conv<<<ceildiv_i(n7, 256), 256, 0, stream>>>(tacc, dWtp, trend_tot);

    // decoder _my_layernorm -> dec
    k_ln1<<<MROWS, 256, 0, stream>>>(dec, dn_g, dn_b, t2);
    k_colmean<<<ceildiv_i(NB * DMODEL, 256), 256, 0, stream>>>(t2, cm);
    k_subcol<<<ceildiv_i((long long)MROWS * 128, 256), 256, 0, stream>>>(
        (const float4*)t2, cm, (float4*)dec);

    // ---- output: trend + seasonal projection, last 256 steps ----
    k_final<<<ceildiv_i((long long)NB * 256 * CIN, 256), 256, 0, stream>>>(
        dec, pW, pb, trend_tot, (float*)d_out);
}
